// ParallelSlowCastle_40046275068466
// MI455X (gfx1250) — compile-verified, matrix operand order checked by reading, not run on hardware
//
#include <hip/hip_runtime.h>
#include <cstdint>
#include <cstddef>

// ---------------------------------------------------------------------------
// ParallelSlowCastle attention for MI455X (gfx1250), wave32 + WMMA bf16.
// B=1, N=2048, DIM=1024, H=16, DH=64.
// GEMMs: LDS double-buffered 128x64 macro-tiles, 32x32 per wave (2x2 WMMA
// accumulators), v_wmma_f32_16x16x32_bf16 with f32 accumulate.
// ---------------------------------------------------------------------------

typedef __bf16  bf16_t;
typedef bf16_t  v16bf __attribute__((ext_vector_type(16)));
typedef float   v8f   __attribute__((ext_vector_type(8)));

#define WMMA_BF16(a, b, c) \
    __builtin_amdgcn_wmma_f32_16x16x32_bf16(false, (a), false, (b), (short)0, (c), false, false)

// LDS tile geometry: K-step 32, rows padded to 40 halfwords (80 B) so the
// per-lane 16 B fragment reads stride 20 banks instead of 16 (kills the
// 4-way stride-64B conflict on the 64-bank LDS).
#define BK     32
#define LDS_LD 40
#define BM     128
#define BN     64

// float -> bf16 round-to-nearest-even (bit pattern as ushort)
__device__ __forceinline__ unsigned short f2bf(float x) {
    unsigned int u = __float_as_uint(x);
    u += 0x7FFFu + ((u >> 16) & 1u);
    return (unsigned short)(u >> 16);
}
__device__ __forceinline__ float bf2f(unsigned short h) {
    return __uint_as_float(((unsigned int)h) << 16);
}

union FragU {
    v16bf v;
    uint4 q[2];
};

// Load one 16x32 (rows x K) bf16 fragment from a row-major tile (global or LDS).
// CDNA5 16-bit A/B layout:
//   lanes 0-15 : row = lane,    halfwords = K {0..7, 16..23}
//   lanes 16-31: row = lane-16, halfwords = K {8..15, 24..31}
// => two 16 B loads per lane at (row, kbase) and (row, kbase+16).
__device__ __forceinline__ v16bf load_frag(const unsigned short* tile, int ld, int lane) {
    int half = lane >> 4;
    int r    = lane & 15;
    const unsigned short* p = tile + (size_t)r * ld + (half ? 8 : 0);
    FragU f;
    f.q[0] = *(const uint4*)(p);
    f.q[1] = *(const uint4*)(p + 16);
    return f.v;
}

// ---------------------------------------------------------------------------
// Core: accumulate a 128x64 macro-tile over K via double-buffered LDS staging.
// Block = (32,8): 8 waves in a 4(M) x 2(N) grid, each wave owns 32x32 (2x2
// accumulators). A: [M,K] row-major from row `tm`; B: [N,K] row-major from
// row `tn` (computing A * B^T). K must be a multiple of 32.
// ---------------------------------------------------------------------------
__device__ __forceinline__ void gemm_accum_128x64(
    const unsigned short* __restrict__ A, int lda,
    const unsigned short* __restrict__ B, int ldb,
    int K,
    unsigned short* As,   // [2][BM*LDS_LD]
    unsigned short* Bs,   // [2][BN*LDS_LD]
    v8f acc[2][2])
{
    const int lane = threadIdx.x;
    const int w    = threadIdx.y;
    const int wm   = w >> 1;      // 0..3
    const int wn   = w & 1;       // 0..1
    const int tid  = w * 32 + lane;

    // Cooperative staging map: one uint4 (8 halfwords) per slot.
    const int ar0 = tid >> 2;             // rows 0..63
    const int ar1 = (tid + 256) >> 2;     // rows 64..127
    const int ac  = (tid & 3) * 8;        // K-chunk within the 32-wide tile
    const int br  = tid >> 2;             // rows 0..63
    const int bc  = (tid & 3) * 8;

    const unsigned short* pa0 = A + (size_t)ar0 * lda + ac;
    const unsigned short* pa1 = A + (size_t)ar1 * lda + ac;
    const unsigned short* pb  = B + (size_t)br  * ldb + bc;

    uint4 ra0 = *(const uint4*)(pa0);
    uint4 ra1 = *(const uint4*)(pa1);
    uint4 rb  = *(const uint4*)(pb);

    int buf = 0;
    for (int k0 = 0; k0 < K; k0 += BK) {
        unsigned short* as = As + buf * (BM * LDS_LD);
        unsigned short* bs = Bs + buf * (BN * LDS_LD);
        *(uint4*)(as + ar0 * LDS_LD + ac) = ra0;
        *(uint4*)(as + ar1 * LDS_LD + ac) = ra1;
        *(uint4*)(bs + br  * LDS_LD + bc) = rb;
        __syncthreads();

        if (k0 + BK < K) {   // issue next-step global loads; consumed next iter
            ra0 = *(const uint4*)(pa0 + k0 + BK);
            ra1 = *(const uint4*)(pa1 + k0 + BK);
            rb  = *(const uint4*)(pb  + k0 + BK);
        }
        if (k0 + 2 * BK < K) {   // global_prefetch_b8 one macro-step further
            __builtin_prefetch(pa0 + k0 + 2 * BK, 0, 1);
            __builtin_prefetch(pb  + k0 + 2 * BK, 0, 1);
        }

        v16bf a0 = load_frag(as + (wm * 32 +  0) * LDS_LD, LDS_LD, lane);
        v16bf a1 = load_frag(as + (wm * 32 + 16) * LDS_LD, LDS_LD, lane);
        v16bf b0 = load_frag(bs + (wn * 32 +  0) * LDS_LD, LDS_LD, lane);
        v16bf b1 = load_frag(bs + (wn * 32 + 16) * LDS_LD, LDS_LD, lane);
        acc[0][0] = WMMA_BF16(a0, b0, acc[0][0]);
        acc[0][1] = WMMA_BF16(a0, b1, acc[0][1]);
        acc[1][0] = WMMA_BF16(a1, b0, acc[1][0]);
        acc[1][1] = WMMA_BF16(a1, b1, acc[1][1]);

        buf ^= 1;
    }
}

// ---------------------------------------------------------------------------
// 0) fp32 -> bf16 conversion
// ---------------------------------------------------------------------------
__global__ void cvt_bf16_kernel(const float* __restrict__ src,
                                unsigned short* __restrict__ dst, int n) {
    int i = blockIdx.x * 256 + threadIdx.x;
    if (i < n) dst[i] = f2bf(src[i]);
}

// ---------------------------------------------------------------------------
// 1) Tiled batched GEMM: C[z] = A[z] (MxK row-major) * B[z]^T (B: NxK row-major)
//    Requires M % 128 == 0, N % 64 == 0, K % 32 == 0 (true for all call sites).
// ---------------------------------------------------------------------------
__global__ void gemm_bf16_tiled(const unsigned short* __restrict__ A,
                                const unsigned short* __restrict__ B,
                                void* __restrict__ Cp,
                                int M, int N, int K,
                                int lda, int ldb, int ldc,
                                long long sA, long long sB, long long sC,
                                int out_bf16) {
    __shared__ unsigned short As[2 * BM * LDS_LD];
    __shared__ unsigned short Bs[2 * BN * LDS_LD];

    int z  = blockIdx.z;
    int tm = blockIdx.y * BM;
    int tn = blockIdx.x * BN;

    const unsigned short* Ab = A + (size_t)z * sA + (size_t)tm * lda;
    const unsigned short* Bb = B + (size_t)z * sB + (size_t)tn * ldb;

    v8f acc[2][2] = {};
    gemm_accum_128x64(Ab, lda, Bb, ldb, K, As, Bs, acc);

    int lane = threadIdx.x, half = lane >> 4;
    int w = threadIdx.y, wm = w >> 1, wn = w & 1;
#pragma unroll
    for (int ii = 0; ii < 2; ii++)
#pragma unroll
        for (int jj = 0; jj < 2; jj++) {
            int rbase = tm + wm * 32 + ii * 16 + half * 8;
            int cbase = tn + wn * 32 + jj * 16 + (lane & 15);
#pragma unroll
            for (int r = 0; r < 8; r++) {
                size_t idx = (size_t)(rbase + r) * ldc + cbase;
                if (out_bf16)
                    ((unsigned short*)Cp + (size_t)z * sC)[idx] = f2bf(acc[ii][jj][r]);
                else
                    ((float*)Cp + (size_t)z * sC)[idx] = acc[ii][jj][r];
            }
        }
}

// ---------------------------------------------------------------------------
// 2) Rearrange qkv [N, 6*H*DH] bf16 -> per-head buffers.
//    qu,qc scaled by DH^-0.5 = 0.125. vc stored transposed [H][DH][N] so the
//    attn@vc GEMM B-operand is K-contiguous.
// ---------------------------------------------------------------------------
__global__ void rearrange_kernel(const unsigned short* __restrict__ qkv,
                                 unsigned short* __restrict__ quS,
                                 unsigned short* __restrict__ ku,
                                 unsigned short* __restrict__ vu,
                                 unsigned short* __restrict__ qcS,
                                 unsigned short* __restrict__ kc,
                                 unsigned short* __restrict__ vcT) {
    const int N = 2048, QKV = 6144;
    int idx = blockIdx.x * 256 + threadIdx.x;
    if (idx >= N * QKV) return;
    int n = idx / QKV;
    int c = idx - n * QKV;
    int q = c / 1024;
    int h = (c % 1024) / 64;
    int d = c % 64;
    unsigned short v = qkv[idx];
    size_t hd = ((size_t)h * N + n) * 64 + d;
    switch (q) {
        case 0: quS[hd] = f2bf(bf2f(v) * 0.125f); break;
        case 1: ku[hd]  = v; break;
        case 2: vu[hd]  = v; break;
        case 3: qcS[hd] = f2bf(bf2f(v) * 0.125f); break;
        case 4: kc[hd]  = v; break;
        case 5: vcT[((size_t)h * 64 + d) * N + n] = v; break;
    }
}

// ---------------------------------------------------------------------------
// 3) Fused term1 + sigmoid-lookahead (K=64, store-bound; simple per-wave tiles)
//    term1[i,j] = (qcS[i].vu[j])        masked to j<=i, stored bf16
//    sig[k,j]   = sigmoid(quS[k].ku[j]) masked to j>k,  stored bf16
// ---------------------------------------------------------------------------
__global__ void term1_sig_kernel(const unsigned short* __restrict__ qcS,
                                 const unsigned short* __restrict__ vu,
                                 const unsigned short* __restrict__ quS,
                                 const unsigned short* __restrict__ ku,
                                 unsigned short* __restrict__ term1,
                                 unsigned short* __restrict__ sigb) {
    const int N = 2048;
    int h  = blockIdx.z;
    int ti = blockIdx.y;
    int tj = blockIdx.x * 4 + threadIdx.y;
    int lane = threadIdx.x, half = lane >> 4;

    size_t hb = (size_t)h * N * 64;
    const unsigned short* Aq = qcS + hb + (size_t)ti * 16 * 64;
    const unsigned short* Bv = vu  + hb + (size_t)tj * 16 * 64;
    const unsigned short* Au = quS + hb + (size_t)ti * 16 * 64;
    const unsigned short* Bk = ku  + hb + (size_t)tj * 16 * 64;

    v8f a1 = {}, a2 = {};
#pragma unroll
    for (int k = 0; k < 64; k += 32) {
        v16bf aq = load_frag(Aq + k, 64, lane);
        v16bf bv = load_frag(Bv + k, 64, lane);
        a1 = WMMA_BF16(aq, bv, a1);
        v16bf au = load_frag(Au + k, 64, lane);
        v16bf bk = load_frag(Bk + k, 64, lane);
        a2 = WMMA_BF16(au, bk, a2);
    }

    size_t base = (size_t)h * N * N;
    int col = tj * 16 + (lane & 15);
#pragma unroll
    for (int r = 0; r < 8; r++) {
        int row = ti * 16 + r + half * 8;
        size_t idx = base + (size_t)row * N + col;
        float t = (col > row) ? 0.0f : a1[r];
        term1[idx] = f2bf(t);
        float s = (col > row) ? (1.0f / (1.0f + __expf(-a2[r]))) : 0.0f;
        sigb[idx] = f2bf(s);
    }
}

// ---------------------------------------------------------------------------
// 4) scores[i,k] = mask(qcS[i].kc[k]) - silu( sum_j term1[i,j]*sig[k,j] )
//    Su: dominant GEMM (K=2048) via LDS-tiled 128x64 path. Sc fused (K=64).
// ---------------------------------------------------------------------------
__global__ void scores_tiled(const unsigned short* __restrict__ term1,
                             const unsigned short* __restrict__ sigb,
                             const unsigned short* __restrict__ qcS,
                             const unsigned short* __restrict__ kc,
                             float* __restrict__ scores) {
    __shared__ unsigned short As[2 * BM * LDS_LD];
    __shared__ unsigned short Bs[2 * BN * LDS_LD];

    const int N = 2048;
    int h  = blockIdx.z;
    int tm = blockIdx.y * BM;   // i rows
    int tn = blockIdx.x * BN;   // k cols
    int lane = threadIdx.x, half = lane >> 4;
    int w = threadIdx.y, wm = w >> 1, wn = w & 1;

    size_t NN = (size_t)N * N;
    const unsigned short* T = term1 + (size_t)h * NN + (size_t)tm * N;
    const unsigned short* S = sigb  + (size_t)h * NN + (size_t)tn * N;

    v8f su[2][2] = {};
    gemm_accum_128x64(T, N, S, N, N, As, Bs, su);

    // Sc = qcS . kc^T for this wave's 32x32 tile, K=64 straight from global.
    size_t hb = (size_t)h * N * 64;
    const unsigned short* Q  = qcS + hb + (size_t)(tm + wm * 32) * 64;
    const unsigned short* Kc = kc  + hb + (size_t)(tn + wn * 32) * 64;
    v8f sc[2][2] = {};
#pragma unroll
    for (int d = 0; d < 64; d += 32) {
        v16bf a0 = load_frag(Q  +            d, 64, lane);
        v16bf a1 = load_frag(Q  + 16 * 64 +  d, 64, lane);
        v16bf b0 = load_frag(Kc +            d, 64, lane);
        v16bf b1 = load_frag(Kc + 16 * 64 +  d, 64, lane);
        sc[0][0] = WMMA_BF16(a0, b0, sc[0][0]);
        sc[0][1] = WMMA_BF16(a0, b1, sc[0][1]);
        sc[1][0] = WMMA_BF16(a1, b0, sc[1][0]);
        sc[1][1] = WMMA_BF16(a1, b1, sc[1][1]);
    }

    float* out = scores + (size_t)h * NN;
#pragma unroll
    for (int ii = 0; ii < 2; ii++)
#pragma unroll
        for (int jj = 0; jj < 2; jj++) {
            int rbase = tm + wm * 32 + ii * 16 + half * 8;
            int cbase = tn + wn * 32 + jj * 16 + (lane & 15);
#pragma unroll
            for (int r = 0; r < 8; r++) {
                int row = rbase + r, col = cbase;
                float x = (col > row) ? -3.402823466e38f : sc[ii][jj][r];
                float u = su[ii][jj][r];
                float sil = u / (1.0f + __expf(-u));
                out[(size_t)row * N + col] = x - sil;
            }
        }
}

// ---------------------------------------------------------------------------
// 5) Row softmax over k (2048 cols), write attn as bf16. LDS tree reduction.
// ---------------------------------------------------------------------------
__global__ void softmax_kernel(const float* __restrict__ scores,
                               unsigned short* __restrict__ attn) {
    const int N = 2048;
    int row = blockIdx.x;
    int h   = blockIdx.y;
    const float* s = scores + ((size_t)h * N + row) * N;
    unsigned short* a = attn + ((size_t)h * N + row) * N;

    __shared__ float red[256];
    int tid = threadIdx.x;

    float vals[8];
    float m = -3.402823466e38f;
#pragma unroll
    for (int i = 0; i < 8; i++) {
        float v = s[tid + i * 256];
        vals[i] = v;
        m = fmaxf(m, v);
    }
    red[tid] = m; __syncthreads();
    for (int st = 128; st > 0; st >>= 1) {
        if (tid < st) red[tid] = fmaxf(red[tid], red[tid + st]);
        __syncthreads();
    }
    m = red[0]; __syncthreads();

    float sum = 0.0f;
#pragma unroll
    for (int i = 0; i < 8; i++) {
        float e = __expf(vals[i] - m);
        vals[i] = e;
        sum += e;
    }
    red[tid] = sum; __syncthreads();
    for (int st = 128; st > 0; st >>= 1) {
        if (tid < st) red[tid] += red[tid + st];
        __syncthreads();
    }
    float inv = 1.0f / red[0];
#pragma unroll
    for (int i = 0; i < 8; i++) a[tid + i * 256] = f2bf(vals[i] * inv);
}

// ---------------------------------------------------------------------------
// Host-side launch
// ---------------------------------------------------------------------------
extern "C" void kernel_launch(void* const* d_in, const int* in_sizes, int n_in,
                              void* d_out, int out_size, void* d_ws, size_t ws_size,
                              hipStream_t stream) {
    const int N = 2048, DIM = 1024, H = 16, DH = 64, QKV = 6 * H * DH; // 6144
    (void)in_sizes; (void)n_in; (void)out_size; (void)ws_size;

    const float* x_f    = (const float*)d_in[0];   // [N, DIM]
    const float* wqkv_f = (const float*)d_in[1];   // [QKV, DIM]
    const float* wout_f = (const float*)d_in[2];   // [DIM, H*DH]
    float* y = (float*)d_out;                      // [N, DIM]

    // Workspace carve-up (256B aligned)
    char* ws = (char*)d_ws;
    size_t off = 0;
    auto take = [&](size_t bytes) -> char* {
        char* p = ws + off;
        off = (off + bytes + 255) & ~(size_t)255;
        return p;
    };
    unsigned short* xb    = (unsigned short*)take((size_t)N * DIM * 2);
    unsigned short* wqkvb = (unsigned short*)take((size_t)QKV * DIM * 2);
    unsigned short* woutb = (unsigned short*)take((size_t)DIM * DIM * 2);
    unsigned short* qkvb  = (unsigned short*)take((size_t)N * QKV * 2);
    unsigned short* quS   = (unsigned short*)take((size_t)H * N * DH * 2);
    unsigned short* ku    = (unsigned short*)take((size_t)H * N * DH * 2);
    unsigned short* vu    = (unsigned short*)take((size_t)H * N * DH * 2);
    unsigned short* qcS   = (unsigned short*)take((size_t)H * N * DH * 2);
    unsigned short* kc    = (unsigned short*)take((size_t)H * N * DH * 2);
    unsigned short* vcT   = (unsigned short*)take((size_t)H * DH * N * 2);
    unsigned short* term1 = (unsigned short*)take((size_t)H * N * N * 2);
    unsigned short* sigb  = (unsigned short*)take((size_t)H * N * N * 2);
    float*          scor  = (float*)take((size_t)H * N * N * 4);
    unsigned short* attnb = (unsigned short*)take((size_t)H * N * N * 2);
    unsigned short* outb  = (unsigned short*)take((size_t)N * (H * DH) * 2);

    dim3 blk(256), wblk(32, 4), tblk(32, 8);

    // 0) fp32 -> bf16
    cvt_bf16_kernel<<<(N * DIM + 255) / 256, blk, 0, stream>>>(x_f, xb, N * DIM);
    cvt_bf16_kernel<<<(QKV * DIM + 255) / 256, blk, 0, stream>>>(wqkv_f, wqkvb, QKV * DIM);
    cvt_bf16_kernel<<<(DIM * DIM + 255) / 256, blk, 0, stream>>>(wout_f, woutb, DIM * DIM);

    // 1) qkv = x @ w_qkv^T  -> bf16 [N, QKV]
    gemm_bf16_tiled<<<dim3(QKV / BN, N / BM, 1), tblk, 0, stream>>>(
        xb, wqkvb, qkvb, N, QKV, DIM, DIM, DIM, QKV, 0, 0, 0, 1);

    // 2) split into per-head buffers (scale qu/qc, transpose vc)
    rearrange_kernel<<<(N * QKV + 255) / 256, blk, 0, stream>>>(
        qkvb, quS, ku, vu, qcS, kc, vcT);

    // 3) term1 + sigmoid lookahead, bf16 [H, N, N] each
    term1_sig_kernel<<<dim3(N / 64, N / 16, H), wblk, 0, stream>>>(
        qcS, vu, quS, ku, term1, sigb);

    // 4) scores = mask(Sc) - silu(Su)   (Su over K=2048: dominant GEMM)
    scores_tiled<<<dim3(N / BN, N / BM, H), tblk, 0, stream>>>(
        term1, sigb, qcS, kc, scor);

    // 5) softmax -> attn bf16
    softmax_kernel<<<dim3(N, H), blk, 0, stream>>>(scor, attnb);

    // 6) out[h] = attn[h] @ vc[h]  -> bf16 [N, H*DH] (batched, col offset h*DH)
    gemm_bf16_tiled<<<dim3(DH / BN, N / BM, H), tblk, 0, stream>>>(
        attnb, vcT, outb, N, DH, N,
        N, N, H * DH,
        (long long)N * N, (long long)DH * N, DH, 1);

    // 7) y = out @ w_out^T -> f32 [N, DIM]
    gemm_bf16_tiled<<<dim3(DIM / BN, N / BM, 1), tblk, 0, stream>>>(
        outb, woutb, y, N, DIM, H * DH, H * DH, H * DH, DIM, 0, 0, 0, 0);
}
